// AttenLayer_60928406061523
// MI455X (gfx1250) — compile-verified, matrix operand order checked by reading, not run on hardware
//
#include <hip/hip_runtime.h>
#include <stdint.h>

// Problem constants (match reference)
#define B_  2
#define N_  2048
#define D_  1024
#define H_  16
#define DH_ 64
#define FF_ 4096
#define ROWS (B_ * N_)          // 4096
#define HD  (H_ * DH_)          // 1024

typedef __attribute__((ext_vector_type(16))) __bf16 v16bf;
typedef __attribute__((ext_vector_type(8)))  float  v8f;

// Fragment: 8 dwords = two 16B chunks; loaded as 2x ds_load_b128.
union FragU { uint4 q[2]; unsigned int u[8]; v16bf v; };

static __device__ __forceinline__ unsigned short f2bf(float f) {
  union { float f; unsigned int u; } c; c.f = f;
  unsigned int u = c.u;
  unsigned int r = u + 0x7FFFu + ((u >> 16) & 1u);   // round-to-nearest-even
  return (unsigned short)(r >> 16);
}

// CDNA5 async global->LDS copy: each lane copies 32B (2 x b128).
// INST_OFFSET is added to BOTH the global and LDS addresses (ISA 08 §4.4).
static __device__ __forceinline__ void async_copy_32B(const void* gsrc, void* ldst) {
  unsigned int       l = (unsigned int)(unsigned long long)ldst;   // low 32 bits = LDS addr
  unsigned long long g = (unsigned long long)gsrc;
  asm volatile("global_load_async_to_lds_b128 %0, %1, off\n\t"
               "global_load_async_to_lds_b128 %0, %1, off offset:16"
               :: "v"(l), "v"(g) : "memory");
}
static __device__ __forceinline__ void async_wait() {
  asm volatile("s_wait_asynccnt 0x0" ::: "memory");
}

// ---------------------------------------------------------------------------
// Elementwise f32 -> bf16 cast (weights), 4 elements/thread
// ---------------------------------------------------------------------------
__global__ void cast_bf16_kernel(const float* __restrict__ in,
                                 unsigned short* __restrict__ out, int n) {
  int i = (blockIdx.x * blockDim.x + threadIdx.x) * 4;
  if (i < n) {
    float4 f = *(const float4*)(in + i);
    union { unsigned short s[4]; uint2 u; } r;
    r.s[0] = f2bf(f.x); r.s[1] = f2bf(f.y); r.s[2] = f2bf(f.z); r.s[3] = f2bf(f.w);
    *(uint2*)(out + i) = r.u;
  }
}

// ---------------------------------------------------------------------------
// Row LayerNorm (D=1024), fp32 in -> bf16 out. One block (256 thr) per row.
// ---------------------------------------------------------------------------
__global__ __launch_bounds__(256)
void layernorm_bf16_kernel(const float* __restrict__ x,
                           const float* __restrict__ g,
                           const float* __restrict__ b,
                           unsigned short* __restrict__ out) {
  __shared__ float red[256];
  int row = blockIdx.x;
  int t = threadIdx.x;
  const float* xr = x + (size_t)row * D_;
  float v[4];
  float s = 0.f;
#pragma unroll
  for (int i = 0; i < 4; ++i) { v[i] = xr[t + i * 256]; s += v[i]; }
  red[t] = s; __syncthreads();
  for (int off = 128; off > 0; off >>= 1) {
    if (t < off) red[t] += red[t + off];
    __syncthreads();
  }
  float mu = red[0] * (1.f / D_);
  __syncthreads();
  float s2 = 0.f;
#pragma unroll
  for (int i = 0; i < 4; ++i) { float d = v[i] - mu; s2 += d * d; }
  red[t] = s2; __syncthreads();
  for (int off = 128; off > 0; off >>= 1) {
    if (t < off) red[t] += red[t + off];
    __syncthreads();
  }
  float rstd = rsqrtf(red[0] * (1.f / D_) + 1e-5f);
#pragma unroll
  for (int i = 0; i < 4; ++i) {
    int c = t + i * 256;
    out[(size_t)row * D_ + c] = f2bf((v[i] - mu) * rstd * g[c] + b[c]);
  }
}

// ---------------------------------------------------------------------------
// Generic bf16 WMMA GEMM:  C[M,N] = A[M,K] @ Bw[K,N]  (f32 accumulate)
// 64x64 tile per block, 128 threads = 4 waves, each wave 16(M) x 64(N).
// Double-buffered software pipeline: async A copy + B global loads for
// tile k+1 are issued before the WMMAs of tile k; B ds_stores after.
// All fragments preloaded so dscnt waits can stagger across the WMMAs.
// ---------------------------------------------------------------------------
__global__ __launch_bounds__(128)
void gemm_wmma_kernel(const unsigned short* __restrict__ A,
                      const unsigned short* __restrict__ Bw,
                      int M, int N, int K,
                      const float* __restrict__ bias, int do_gelu,
                      const float* __restrict__ resid,
                      unsigned short* __restrict__ outb,
                      float* __restrict__ outf) {
  __shared__ alignas(16) unsigned short Alds[2][64][40];  // [m][k], 80B rows (padded)
  __shared__ alignas(16) unsigned int   Bt[2][64][20];    // [n][kpair], 80B rows (padded)

  int t = threadIdx.x;
  int wave = t >> 5, lane = t & 31;
  int lm = lane & 15, hf = lane >> 4;
  int mBase = blockIdx.y * 64;
  int nBase = blockIdx.x * 64;
  int rowBase = wave * 16;
  int kIters = K >> 5;

  auto stageA = [&](int kt, int buf) {
    int row = t >> 1, c16 = (t & 1) * 16;
    async_copy_32B(A + (size_t)(mBase + row) * K + kt * 32 + c16,
                   &Alds[buf][row][c16]);
  };
  unsigned int br0[4], br1[4];
  auto loadB = [&](int kt) {
#pragma unroll
    for (int i = 0; i < 4; ++i) {
      int idx = i * 128 + t;           // 512 tasks: (kp, n-pair)
      int kp = idx >> 5;               // 0..15
      int n0 = (idx & 31) * 2;         // 0..62 even
      int krow = kt * 32 + kp * 2;
      br0[i] = *(const unsigned int*)&Bw[(size_t)krow * N + nBase + n0];
      br1[i] = *(const unsigned int*)&Bw[(size_t)(krow + 1) * N + nBase + n0];
      if (kt + 1 < kIters)
        __builtin_prefetch(&Bw[(size_t)(krow + 32) * N + nBase + n0], 0, 1);
    }
  };
  auto storeB = [&](int buf) {
#pragma unroll
    for (int i = 0; i < 4; ++i) {
      int idx = i * 128 + t;
      int kp = idx >> 5;
      int n0 = (idx & 31) * 2;
      Bt[buf][n0][kp]     = (br0[i] & 0xFFFFu) | (br1[i] << 16);
      Bt[buf][n0 + 1][kp] = (br0[i] >> 16)     | (br1[i] & 0xFFFF0000u);
    }
  };

  v8f vzero = {};
  v8f acc[4];
#pragma unroll
  for (int c = 0; c < 4; ++c) acc[c] = vzero;

  // prologue: stage tile 0
  stageA(0, 0);
  loadB(0);
  storeB(0);
  async_wait();
  __syncthreads();

  for (int kt = 0; kt < kIters; ++kt) {
    int cur = kt & 1;
    bool more = (kt + 1) < kIters;
    if (more) { stageA(kt + 1, cur ^ 1); loadB(kt + 1); }

    // ---- preload all fragments (2x b128 each), then 4 WMMAs ----
    FragU af, bf[4];
    int arow = rowBase + lm;
    af.q[0] = *(const uint4*)&Alds[cur][arow][hf * 8];
    af.q[1] = *(const uint4*)&Alds[cur][arow][16 + hf * 8];
#pragma unroll
    for (int c = 0; c < 4; ++c) {
      bf[c].q[0] = *(const uint4*)&Bt[cur][c * 16 + lm][hf * 4];
      bf[c].q[1] = *(const uint4*)&Bt[cur][c * 16 + lm][8 + hf * 4];
    }
#pragma unroll
    for (int c = 0; c < 4; ++c) {
      acc[c] = __builtin_amdgcn_wmma_f32_16x16x32_bf16(
          false, af.v, false, bf[c].v, (short)0, acc[c], false, false);
    }

    if (more) storeB(cur ^ 1);
    async_wait();
    __syncthreads();
  }

  // ---- epilogue ----
#pragma unroll
  for (int c = 0; c < 4; ++c) {
    int col = nBase + c * 16 + lm;
#pragma unroll
    for (int v = 0; v < 8; ++v) {
      int row = mBase + rowBase + v + hf * 8;
      float val = acc[c][v];
      if (bias) val += bias[col];
      if (do_gelu) val = 0.5f * val * (1.f + erff(val * 0.70710678118654752f));
      if (resid) val += resid[(size_t)row * N + col];
      if (outf) outf[(size_t)row * N + col] = val;
      else      outb[(size_t)row * N + col] = f2bf(val);
    }
  }
}

// ---------------------------------------------------------------------------
// Flash attention (no 1/sqrt(d) scaling, per reference).
// Q/K/V bf16 laid out [B*N, H*DH]. Grid: (B*H, N/64), block 128 (4 waves).
// Double-buffered K (async) and V (regs->packed LDS) tiles.
// ---------------------------------------------------------------------------
__global__ __launch_bounds__(128)
void flash_attn_kernel(const unsigned short* __restrict__ Q,
                       const unsigned short* __restrict__ Km,
                       const unsigned short* __restrict__ Vm,
                       unsigned short* __restrict__ O) {
  __shared__ alignas(16) unsigned short Qs[64][72];       // 144B rows
  __shared__ alignas(16) unsigned short Ks[2][32][72];    // 144B rows
  __shared__ alignas(16) unsigned int   Vt[2][64][20];    // [dh][keypair], 80B rows
  __shared__ alignas(16) unsigned short Ps[4][16][40];    // per-wave P, 80B rows

  int t = threadIdx.x;
  int wave = t >> 5, lane = t & 31;
  int lm = lane & 15, hf = lane >> 4;
  int bh = blockIdx.x;
  int b = bh / H_, h = bh % H_;
  int q0 = b * N_ + blockIdx.y * 64;
  int colh = h * DH_;
  int rowBase = wave * 16;
  const int kbIters = N_ / 32;

  auto stageK = [&](int kb, int buf) {
    int k0 = b * N_ + kb * 32;
    int row = t >> 2, c16 = (t & 3) * 16;
    async_copy_32B(Km + (size_t)(k0 + row) * HD + colh + c16, &Ks[buf][row][c16]);
  };
  unsigned int vr0[4], vr1[4];
  auto loadV = [&](int kb) {
    int k0 = b * N_ + kb * 32;
#pragma unroll
    for (int i = 0; i < 4; ++i) {
      int idx = i * 128 + t;           // 512 tasks: (kp, dh-pair)
      int kp  = idx >> 5;              // 0..15
      int dh0 = (idx & 31) * 2;        // 0..62 even
      vr0[i] = *(const unsigned int*)&Vm[(size_t)(k0 + 2 * kp) * HD + colh + dh0];
      vr1[i] = *(const unsigned int*)&Vm[(size_t)(k0 + 2 * kp + 1) * HD + colh + dh0];
    }
  };
  auto storeV = [&](int buf) {
#pragma unroll
    for (int i = 0; i < 4; ++i) {
      int idx = i * 128 + t;
      int kp  = idx >> 5;
      int dh0 = (idx & 31) * 2;
      Vt[buf][dh0][kp]     = (vr0[i] & 0xFFFFu) | (vr1[i] << 16);
      Vt[buf][dh0 + 1][kp] = (vr0[i] >> 16)     | (vr1[i] & 0xFFFF0000u);
    }
  };

  // load Q tile 64x64 (64B per thread)
  {
    int row = t >> 1, c32 = (t & 1) * 32;
    const uint4* src = (const uint4*)(Q + (size_t)(q0 + row) * HD + colh + c32);
    uint4* dst = (uint4*)&Qs[row][c32];
    dst[0] = src[0]; dst[1] = src[1]; dst[2] = src[2]; dst[3] = src[3];
  }

  v8f vzero = {};
  v8f o[4];
#pragma unroll
  for (int c = 0; c < 4; ++c) o[c] = vzero;
  float mrow[8], lrow[8];
#pragma unroll
  for (int v = 0; v < 8; ++v) { mrow[v] = -1e30f; lrow[v] = 0.f; }

  // prologue: stage tile 0 (barrier also publishes Qs)
  stageK(0, 0);
  loadV(0);
  storeV(0);
  async_wait();
  __syncthreads();

  for (int kb = 0; kb < kbIters; ++kb) {
    int cur = kb & 1;
    bool more = (kb + 1) < kbIters;
    if (more) { stageK(kb + 1, cur ^ 1); loadV(kb + 1); }

    // ---- S = Q K^T  (two 16-key tiles, two k-steps over DH=64) ----
    v8f s[2]; s[0] = vzero; s[1] = vzero;
#pragma unroll
    for (int step = 0; step < 64; step += 32) {
      FragU qa, kf0, kf1;
      int arow = rowBase + lm;
      qa.q[0]  = *(const uint4*)&Qs[arow][step + hf * 8];
      qa.q[1]  = *(const uint4*)&Qs[arow][step + 16 + hf * 8];
      kf0.q[0] = *(const uint4*)&Ks[cur][lm][step + hf * 8];
      kf0.q[1] = *(const uint4*)&Ks[cur][lm][step + 16 + hf * 8];
      kf1.q[0] = *(const uint4*)&Ks[cur][16 + lm][step + hf * 8];
      kf1.q[1] = *(const uint4*)&Ks[cur][16 + lm][step + 16 + hf * 8];
      s[0] = __builtin_amdgcn_wmma_f32_16x16x32_bf16(
          false, qa.v, false, kf0.v, (short)0, s[0], false, false);
      s[1] = __builtin_amdgcn_wmma_f32_16x16x32_bf16(
          false, qa.v, false, kf1.v, (short)0, s[1], false, false);
    }

    // ---- online softmax (row stats live in WMMA C layout: row = v + hf*8) ----
    float p0[8], p1[8];
#pragma unroll
    for (int v = 0; v < 8; ++v) {
      float tm = fmaxf(s[0][v], s[1][v]);
#pragma unroll
      for (int msk = 8; msk >= 1; msk >>= 1) tm = fmaxf(tm, __shfl_xor(tm, msk, 32));
      float mn = fmaxf(mrow[v], tm);
      float scale = __expf(mrow[v] - mn);
      mrow[v] = mn;
      p0[v] = __expf(s[0][v] - mn);
      p1[v] = __expf(s[1][v] - mn);
      float rs = p0[v] + p1[v];
#pragma unroll
      for (int msk = 8; msk >= 1; msk >>= 1) rs += __shfl_xor(rs, msk, 32);
      lrow[v] = lrow[v] * scale + rs;
#pragma unroll
      for (int c = 0; c < 4; ++c) o[c][v] *= scale;
    }

    // ---- round-trip P through LDS to get A-fragment layout ----
#pragma unroll
    for (int v = 0; v < 8; ++v) {
      int pr = v + hf * 8;
      Ps[wave][pr][lm]      = f2bf(p0[v]);
      Ps[wave][pr][16 + lm] = f2bf(p1[v]);
    }
    __syncthreads();

    // ---- O += P @ V (preload all fragments, then 4 WMMAs) ----
    FragU pa, vf[4];
    pa.q[0] = *(const uint4*)&Ps[wave][lm][hf * 8];
    pa.q[1] = *(const uint4*)&Ps[wave][lm][16 + hf * 8];
#pragma unroll
    for (int ct = 0; ct < 4; ++ct) {
      vf[ct].q[0] = *(const uint4*)&Vt[cur][ct * 16 + lm][hf * 4];
      vf[ct].q[1] = *(const uint4*)&Vt[cur][ct * 16 + lm][8 + hf * 4];
    }
#pragma unroll
    for (int ct = 0; ct < 4; ++ct) {
      o[ct] = __builtin_amdgcn_wmma_f32_16x16x32_bf16(
          false, pa.v, false, vf[ct].v, (short)0, o[ct], false, false);
    }

    if (more) storeV(cur ^ 1);
    async_wait();
    __syncthreads();
  }

  // ---- normalize and store bf16 ----
#pragma unroll
  for (int ct = 0; ct < 4; ++ct) {
#pragma unroll
    for (int v = 0; v < 8; ++v) {
      int row = q0 + rowBase + v + hf * 8;
      int col = colh + ct * 16 + lm;
      O[(size_t)row * HD + col] = f2bf(o[ct][v] / lrow[v]);
    }
  }
}

// ---------------------------------------------------------------------------
// Host-side orchestration
// ---------------------------------------------------------------------------
extern "C" void kernel_launch(void* const* d_in, const int* in_sizes, int n_in,
                              void* d_out, int out_size, void* d_ws, size_t ws_size,
                              hipStream_t stream) {
  const float* x     = (const float*)d_in[0];
  const float* wq    = (const float*)d_in[1];
  const float* wk    = (const float*)d_in[2];
  const float* wv    = (const float*)d_in[3];
  const float* wo    = (const float*)d_in[4];
  const float* w1    = (const float*)d_in[5];
  const float* b1    = (const float*)d_in[6];
  const float* w2    = (const float*)d_in[7];
  const float* b2    = (const float*)d_in[8];
  const float* ln1_g = (const float*)d_in[9];
  const float* ln1_b = (const float*)d_in[10];
  const float* ln2_g = (const float*)d_in[11];
  const float* ln2_b = (const float*)d_in[12];
  float* out = (float*)d_out;

  // workspace carve-out (~120 MiB total)
  char* p = (char*)d_ws;
  auto alloc = [&](size_t bytes) -> void* {
    void* r = (void*)p;
    p += (bytes + 255) & ~(size_t)255;
    return r;
  };
  unsigned short* wq_b = (unsigned short*)alloc((size_t)D_ * HD * 2);
  unsigned short* wk_b = (unsigned short*)alloc((size_t)D_ * HD * 2);
  unsigned short* wv_b = (unsigned short*)alloc((size_t)D_ * HD * 2);
  unsigned short* wo_b = (unsigned short*)alloc((size_t)HD * D_ * 2);
  unsigned short* w1_b = (unsigned short*)alloc((size_t)D_ * FF_ * 2);
  unsigned short* w2_b = (unsigned short*)alloc((size_t)FF_ * D_ * 2);
  unsigned short* h_b  = (unsigned short*)alloc((size_t)ROWS * D_ * 2);
  unsigned short* q_b  = (unsigned short*)alloc((size_t)ROWS * HD * 2);
  unsigned short* k_b  = (unsigned short*)alloc((size_t)ROWS * HD * 2);
  unsigned short* v_b  = (unsigned short*)alloc((size_t)ROWS * HD * 2);
  unsigned short* a_b  = (unsigned short*)alloc((size_t)ROWS * HD * 2);
  float*          x2   = (float*)        alloc((size_t)ROWS * D_ * 4);
  unsigned short* h2_b = (unsigned short*)alloc((size_t)ROWS * D_ * 2);
  unsigned short* ff_b = (unsigned short*)alloc((size_t)ROWS * FF_ * 2);

  // 1) weights -> bf16
  auto cast = [&](const float* src, unsigned short* dst, int n) {
    cast_bf16_kernel<<<(n / 4 + 255) / 256, 256, 0, stream>>>(src, dst, n);
  };
  cast(wq, wq_b, D_ * HD);
  cast(wk, wk_b, D_ * HD);
  cast(wv, wv_b, D_ * HD);
  cast(wo, wo_b, HD * D_);
  cast(w1, w1_b, D_ * FF_);
  cast(w2, w2_b, FF_ * D_);

  // 2) LN1
  layernorm_bf16_kernel<<<ROWS, 256, 0, stream>>>(x, ln1_g, ln1_b, h_b);

  // 3) Q/K/V projections
  dim3 gemmBlk(128);
  dim3 gQKV(D_ / 64, ROWS / 64);
  gemm_wmma_kernel<<<gQKV, gemmBlk, 0, stream>>>(h_b, wq_b, ROWS, HD, D_,
                                                 nullptr, 0, nullptr, q_b, nullptr);
  gemm_wmma_kernel<<<gQKV, gemmBlk, 0, stream>>>(h_b, wk_b, ROWS, HD, D_,
                                                 nullptr, 0, nullptr, k_b, nullptr);
  gemm_wmma_kernel<<<gQKV, gemmBlk, 0, stream>>>(h_b, wv_b, ROWS, HD, D_,
                                                 nullptr, 0, nullptr, v_b, nullptr);

  // 4) attention
  dim3 gAttn(B_ * H_, N_ / 64);
  flash_attn_kernel<<<gAttn, gemmBlk, 0, stream>>>(q_b, k_b, v_b, a_b);

  // 5) out projection + residual -> x2 (fp32)
  gemm_wmma_kernel<<<gQKV, gemmBlk, 0, stream>>>(a_b, wo_b, ROWS, D_, HD,
                                                 nullptr, 0, x, nullptr, x2);

  // 6) LN2
  layernorm_bf16_kernel<<<ROWS, 256, 0, stream>>>(x2, ln2_g, ln2_b, h2_b);

  // 7) FF1 + bias + exact GELU -> bf16
  dim3 gFF1(FF_ / 64, ROWS / 64);
  gemm_wmma_kernel<<<gFF1, gemmBlk, 0, stream>>>(h2_b, w1_b, ROWS, FF_, D_,
                                                 b1, 1, nullptr, ff_b, nullptr);

  // 8) FF2 + bias + residual -> out (fp32)
  dim3 gFF2(D_ / 64, ROWS / 64);
  gemm_wmma_kernel<<<gFF2, gemmBlk, 0, stream>>>(ff_b, w2_b, ROWS, D_, FF_,
                                                 b2, 0, x2, nullptr, out);
}
